// Backflow_35536559407454
// MI455X (gfx1250) — matrix operand order repaired and checked by reading, not compiled, not run on hardware
//
#include <hip/hip_runtime.h>
#include <math.h>

typedef __attribute__((ext_vector_type(16))) _Float16 v16h;
typedef __attribute__((ext_vector_type(8)))  float    v8f;

__device__ __forceinline__ float softplus_f(float z) {
    // Raw-hardware softplus: max(z,0) + ln2 * log2(1 + 2^(-log2e*|z|)).
    // Argument of v_log_f32 is in [1,2]: never denormal/inf/NaN, so the
    // OCML edge-case fixups (denorm rescale, inf select, compensated ln2
    // conversion) are provably unnecessary. ~6 VALU ops total.
    float p = __builtin_amdgcn_exp2f(-1.442695040888963f * fabsf(z)); // exp(-|z|)
    float l = __builtin_amdgcn_logf(1.0f + p);                        // log2(1+p)
    return fmaf(0.6931471805599453f, l, fmaxf(z, 0.0f));
}

// ---------------------------------------------------------------------------
// One-body term: out[i] = xi(|x_i|, t) * x_i   (initializes d_out)
// Negligible cost (1024 MLP evals) -> plain scalar VALU kernel.
// ---------------------------------------------------------------------------
__global__ __launch_bounds__(256) void backflow_onebody(
    const float* __restrict__ x, const float* __restrict__ tptr,
    const float* __restrict__ W1, const float* __restrict__ b1,
    const float* __restrict__ W2, const float* __restrict__ b2,
    const float* __restrict__ W3, const float* __restrict__ b3,
    float* __restrict__ out, int n)
{
    int i = blockIdx.x * blockDim.x + threadIdx.x;
    if (i >= n) return;
    float xx = x[3*i+0], yy = x[3*i+1], zz = x[3*i+2];
    float r = sqrtf(xx*xx + yy*yy + zz*zz);
    float t = tptr[0];

    float h1[64];
    #pragma unroll
    for (int k = 0; k < 64; ++k)
        h1[k] = softplus_f(fmaf(r, W1[k], fmaf(t, W1[64 + k], b1[k])));

    float o = b3[0];
    for (int k2 = 0; k2 < 64; ++k2) {
        float s = b2[k2];
        #pragma unroll
        for (int k = 0; k < 64; ++k)
            s = fmaf(h1[k], W2[k * 64 + k2], s);
        o = fmaf(softplus_f(s), W3[k2], o);
    }
    out[3*i+0] = o * xx;
    out[3*i+1] = o * yy;
    out[3*i+2] = o * zz;
}

// ---------------------------------------------------------------------------
// Two-body term: out[i] += sum_j eta(|x_i-x_j|, t) * (x_i - x_j)
// One wave32 per i. Layer-2 (64x64) runs on v_wmma_f32_16x16x32_f16 with
// loop-invariant W2 B-fragments kept in registers.
// ---------------------------------------------------------------------------
__global__ __launch_bounds__(256) void backflow_twobody(
    const float* __restrict__ x, const float* __restrict__ tptr,
    const float* __restrict__ eW1, const float* __restrict__ eb1,
    const float* __restrict__ eW2, const float* __restrict__ eb2,
    const float* __restrict__ eW3, const float* __restrict__ eb3,
    float* __restrict__ out, int n)
{
    const int lane = threadIdx.x & 31;
    const int wave = blockIdx.x * (blockDim.x >> 5) + (threadIdx.x >> 5);
    const int i = wave;                 // wave-uniform guard: EXEC stays all-1
    if (i >= n) return;

    const int nn = lane & 15;           // tile row/col id (M for A, N for B/C)
    const int g  = lane >> 4;           // half-wave id
    const float t = tptr[0];
    const float xix = x[3*i+0], xiy = x[3*i+1], xiz = x[3*i+2];

    // Per-lane layer-1 weights matching the A-fragment K mapping:
    // element e of kstep ks -> K = 32*ks + 16*(e>>3) + 8*g + (e&7)
    float w1a[32], w1c[32];
    #pragma unroll
    for (int kk = 0; kk < 32; ++kk) {
        int e = kk & 15, ks = kk >> 4;
        int K = 32*ks + 16*(e >> 3) + 8*g + (e & 7);
        w1a[kk] = eW1[K];
        w1c[kk] = fmaf(t, eW1[64 + K], eb1[K]);
    }

    // Loop-invariant B fragments of W2 (64x64), f16.
    // B (32x16) fragment: lane (nn,g), element e -> W2[K = 32*ks+16*(e>>3)+8*g+(e&7)][16*tile+nn]
    v16h bfrag[2][4];
    #pragma unroll
    for (int ks = 0; ks < 2; ++ks)
        #pragma unroll
        for (int tile = 0; tile < 4; ++tile)
            #pragma unroll
            for (int e = 0; e < 16; ++e) {
                int K = 32*ks + 16*(e >> 3) + 8*g + (e & 7);
                bfrag[ks][tile][e] = (_Float16)eW2[K * 64 + 16*tile + nn];
            }

    float b2l[4], w3l[4];
    #pragma unroll
    for (int tile = 0; tile < 4; ++tile) {
        b2l[tile] = eb2[16*tile + nn];
        w3l[tile] = eW3[16*tile + nn];
    }
    const float b3 = eb3[0];

    float accx = 0.f, accy = 0.f, accz = 0.f;

    for (int j0 = 0; j0 < n; j0 += 16) {
        // distance for pair row m = nn (computed redundantly in both halves)
        int j = j0 + nn; if (j >= n) j = n - 1;
        float dx = xix - x[3*j+0];
        float dy = xiy - x[3*j+1];
        float dz = xiz - x[3*j+2];
        float r  = sqrtf(fmaf(dx, dx, fmaf(dy, dy, dz*dz)));  // ==0 when j==i

        // C init with layer-2 bias (per column N = 16*tile + nn)
        v8f c[4];
        #pragma unroll
        for (int tile = 0; tile < 4; ++tile)
            #pragma unroll
            for (int rr = 0; rr < 8; ++rr)
                c[tile][rr] = b2l[tile];

        // layer 1 (softplus) -> A fragment; layer 2 on WMMA
        #pragma unroll
        for (int ks = 0; ks < 2; ++ks) {
            v16h a;
            #pragma unroll
            for (int e = 0; e < 16; ++e) {
                float z = fmaf(r, w1a[ks*16 + e], w1c[ks*16 + e]);
                a[e] = (_Float16)softplus_f(z);
            }
            #pragma unroll
            for (int tile = 0; tile < 4; ++tile)
                c[tile] = __builtin_amdgcn_wmma_f32_16x16x32_f16(
                    false, a, false, bfrag[ks][tile],
                    (short)0, c[tile], false, false);
        }

        // layer 3: softplus(h2) . W3, reduced across the 16 columns per half.
        // Row M = rr + 8*g lives in VGPR rr of half g.
        float part[8];
        #pragma unroll
        for (int rr = 0; rr < 8; ++rr) {
            float s = 0.f;
            #pragma unroll
            for (int tile = 0; tile < 4; ++tile)
                s = fmaf(softplus_f(c[tile][rr]), w3l[tile], s);
            s += __shfl_xor(s, 1, 32);
            s += __shfl_xor(s, 2, 32);
            s += __shfl_xor(s, 4, 32);
            s += __shfl_xor(s, 8, 32);   // all 16 lanes of each half hold sum
            part[rr] = s;
        }

        // lanes nn<8 each own one row m2 = 8*g + nn -> accumulate eta*rij
        if (nn < 8) {
            float eta = part[0];
            #pragma unroll
            for (int rr = 1; rr < 8; ++rr)
                eta = (nn == rr) ? part[rr] : eta;
            eta += b3;
            int m2 = 8*g + nn;
            int j2 = j0 + m2;
            if (j2 < n) {
                float ddx = xix - x[3*j2+0];
                float ddy = xiy - x[3*j2+1];
                float ddz = xiz - x[3*j2+2];
                accx = fmaf(eta, ddx, accx);
                accy = fmaf(eta, ddy, accy);
                accz = fmaf(eta, ddz, accz);
            }
        }
    }

    // wave-wide reduction of the per-row accumulators
    #pragma unroll
    for (int m = 16; m >= 1; m >>= 1) {
        accx += __shfl_xor(accx, m, 32);
        accy += __shfl_xor(accy, m, 32);
        accz += __shfl_xor(accz, m, 32);
    }
    if (lane == 0) {
        out[3*i+0] += accx;
        out[3*i+1] += accy;
        out[3*i+2] += accz;
    }
}

// ---------------------------------------------------------------------------
extern "C" void kernel_launch(void* const* d_in, const int* in_sizes, int n_in,
                              void* d_out, int out_size, void* d_ws, size_t ws_size,
                              hipStream_t stream) {
    const float* x     = (const float*)d_in[0];
    const float* t     = (const float*)d_in[1];
    const float* xiW1  = (const float*)d_in[2];
    const float* xib1  = (const float*)d_in[3];
    const float* xiW2  = (const float*)d_in[4];
    const float* xib2  = (const float*)d_in[5];
    const float* xiW3  = (const float*)d_in[6];
    const float* xib3  = (const float*)d_in[7];
    const float* eW1   = (const float*)d_in[8];
    const float* eb1   = (const float*)d_in[9];
    const float* eW2   = (const float*)d_in[10];
    const float* eb2   = (const float*)d_in[11];
    const float* eW3   = (const float*)d_in[12];
    const float* eb3   = (const float*)d_in[13];
    float* out = (float*)d_out;
    const int n = in_sizes[0] / 3;

    // 1) one-body term initializes out
    backflow_onebody<<<(n + 255) / 256, 256, 0, stream>>>(
        x, t, xiW1, xib1, xiW2, xib2, xiW3, xib3, out, n);

    // 2) two-body term: one wave32 per particle i, 8 waves per block
    const int waves_per_block = 8;
    const int nblocks = (n + waves_per_block - 1) / waves_per_block;
    backflow_twobody<<<nblocks, waves_per_block * 32, 0, stream>>>(
        x, t, eW1, eb1, eW2, eb2, eW3, eb3, out, n);
}
